// CAWN2_70961449664996
// MI455X (gfx1250) — compile-verified, matrix operand order checked by reading, not run on hardware
//
#include <hip/hip_runtime.h>

// ---------------------------------------------------------------------------
// CDNA5 (gfx1250) types & helpers
// ---------------------------------------------------------------------------
typedef __attribute__((ext_vector_type(16))) __bf16 v16bf;
typedef __attribute__((ext_vector_type(8)))  float  v8f;

union FragU { v16bf v; uint4 q[2]; };

// fp32 -> bf16 bits, round-to-nearest-even
__device__ __forceinline__ unsigned f2bf1(float f) {
  union { float f; unsigned u; } a; a.f = f;
  unsigned u = a.u;
  u += 0x7fffu + ((u >> 16) & 1u);
  return u >> 16;
}
__device__ __forceinline__ unsigned pk2(float lo, float hi) {
  return f2bf1(lo) | (f2bf1(hi) << 16);
}
// 8 consecutive fp32 -> 8 bf16 packed in a uint4 (K ascending, even in [15:0])
__device__ __forceinline__ uint4 cvt8(const float* p) {
  float4 a = *(const float4*)p;
  float4 b = *(const float4*)(p + 4);
  uint4 r;
  r.x = pk2(a.x, a.y); r.y = pk2(a.z, a.w);
  r.z = pk2(b.x, b.y); r.w = pk2(b.z, b.w);
  return r;
}

// A-fragment (16x32 bf16, MxK) loaded from row-major fp32 global memory.
// Lane L: row = rbase + (L&15); K chunks at kf*32 + (L>=16 ? 8 : 0) and +16.
__device__ __forceinline__ v16bf afrag(const float* __restrict__ X,
                                       int rbase, int kf, int lane) {
  const float* p = X + (size_t)(rbase + (lane & 15)) * 128;
  int c0 = kf * 32 + ((lane >> 4) << 3);
  FragU f;
  f.q[0] = cvt8(p + c0);
  f.q[1] = cvt8(p + c0 + 16);
  return f.v;
}

// B-fragment (32x16 bf16, KxN) for D = A x B with B[k,n] = W[n,k]:
// mirrored layout (lane = column n), read from padded bf16 LDS (row stride 136).
__device__ __forceinline__ v16bf bfrag_lds(const unsigned short* sW,
                                           int n0, int kf, int lane) {
  int e = (n0 + (lane & 15)) * 136 + kf * 32 + ((lane >> 4) << 3);
  FragU f;
  f.q[0] = *(const uint4*)(sW + e);
  f.q[1] = *(const uint4*)(sW + e + 16);
  return f.v;
}

__device__ __forceinline__ v8f wmma_bf16(v16bf a, v16bf b, v8f c) {
  return __builtin_amdgcn_wmma_f32_16x16x32_bf16(
      /*neg_a=*/false, a, /*neg_b=*/false, b,
      /*c_mod=*/(short)0, c, /*reuse_a=*/false, /*reuse_b=*/false);
}

#define MROWS (256 * 2048)       // B * N_NGH
#define NTILES (MROWS / 16)      // 32768 16-row tiles

// ---------------------------------------------------------------------------
// Kernel 1: fused GRUCell + K/V projection.
//   seq = GRU(ngh_feat, hidden);  K = seq @ Wks^T;  V = seq @ Wvs^T
// Weights resident in LDS as bf16 (padded rows); seq staged per-wave in LDS.
// ---------------------------------------------------------------------------
__global__ __launch_bounds__(256) void gru_kv_kernel(
    const float* __restrict__ x,   const float* __restrict__ h,
    const float* __restrict__ Wih, const float* __restrict__ Whh,
    const float* __restrict__ bih, const float* __restrict__ bhh,
    const float* __restrict__ Wks, const float* __restrict__ Wvs,
    float* __restrict__ Kout,      float* __restrict__ Vout) {
  __shared__ __align__(16) unsigned short sWih[384 * 136];
  __shared__ __align__(16) unsigned short sWhh[384 * 136];
  __shared__ __align__(16) unsigned short sWks[128 * 136];
  __shared__ __align__(16) unsigned short sWvs[128 * 136];
  __shared__ __align__(16) unsigned short sStage[8 * 16 * 136];

  // Cooperative weight load + fp32->bf16 convert, once per block.
  for (int i = threadIdx.x; i < 384 * 128; i += blockDim.x) {
    int r = i >> 7, c = i & 127;
    sWih[r * 136 + c] = (unsigned short)f2bf1(Wih[i]);
    sWhh[r * 136 + c] = (unsigned short)f2bf1(Whh[i]);
  }
  for (int i = threadIdx.x; i < 128 * 128; i += blockDim.x) {
    int r = i >> 7, c = i & 127;
    sWks[r * 136 + c] = (unsigned short)f2bf1(Wks[i]);
    sWvs[r * 136 + c] = (unsigned short)f2bf1(Wvs[i]);
  }
  __syncthreads();

  const int lane = threadIdx.x & 31;
  const int wave = threadIdx.x >> 5;
  unsigned short* stg = sStage + wave * (16 * 136);
  const v8f zf = {0.f, 0.f, 0.f, 0.f, 0.f, 0.f, 0.f, 0.f};

  for (int tile = blockIdx.x * 8 + wave; tile < NTILES; tile += gridDim.x * 8) {
    const int rbase = tile * 16;

    v16bf xa[4], ha[4];
#pragma unroll
    for (int kf = 0; kf < 4; ++kf) {
      xa[kf] = afrag(x, rbase, kf, lane);
      ha[kf] = afrag(h, rbase, kf, lane);
    }

    // ---- gate GEMMs + elementwise GRU math, 16-column block at a time ----
    for (int cb = 0; cb < 8; ++cb) {
      const int n0 = cb * 16;
      v8f aIR = zf, aIZ = zf, aIN = zf, aHR = zf, aHZ = zf, aHN = zf;
#pragma unroll
      for (int kf = 0; kf < 4; ++kf) {
        aIR = wmma_bf16(xa[kf], bfrag_lds(sWih, n0,       kf, lane), aIR);
        aIZ = wmma_bf16(xa[kf], bfrag_lds(sWih, n0 + 128, kf, lane), aIZ);
        aIN = wmma_bf16(xa[kf], bfrag_lds(sWih, n0 + 256, kf, lane), aIN);
        aHR = wmma_bf16(ha[kf], bfrag_lds(sWhh, n0,       kf, lane), aHR);
        aHZ = wmma_bf16(ha[kf], bfrag_lds(sWhh, n0 + 128, kf, lane), aHZ);
        aHN = wmma_bf16(ha[kf], bfrag_lds(sWhh, n0 + 256, kf, lane), aHN);
      }
      const int col  = n0 + (lane & 15);
      const int rofs = (lane >> 4) << 3;
      const float bR = bih[col], bZ = bih[col + 128], bN = bih[col + 256];
      const float cR = bhh[col], cZ = bhh[col + 128], cN = bhh[col + 256];
#pragma unroll
      for (int r = 0; r < 8; ++r) {
        const int grow = rbase + r + rofs;
        float ir = aIR[r] + bR, iz = aIZ[r] + bZ, inn = aIN[r] + bN;
        float hr = aHR[r] + cR, hz = aHZ[r] + cZ, hn  = aHN[r] + cN;
        float rg = 1.f / (1.f + __expf(-(ir + hr)));
        float zg = 1.f / (1.f + __expf(-(iz + hz)));
        float ng = tanhf(inn + rg * hn);
        float hv = h[(size_t)grow * 128 + col];
        float sv = (1.f - zg) * ng + zg * hv;
        stg[(r + rofs) * 136 + col] = (unsigned short)f2bf1(sv);
      }
    }

    // ---- K/V projection from staged seq tile (same-wave DS is in-order) ----
    v16bf sa[4];
#pragma unroll
    for (int kf = 0; kf < 4; ++kf) sa[kf] = bfrag_lds(stg, 0, kf, lane);

    for (int cb = 0; cb < 8; ++cb) {
      v8f aK = zf, aV = zf;
#pragma unroll
      for (int kf = 0; kf < 4; ++kf) {
        aK = wmma_bf16(sa[kf], bfrag_lds(sWks, cb * 16, kf, lane), aK);
        aV = wmma_bf16(sa[kf], bfrag_lds(sWvs, cb * 16, kf, lane), aV);
      }
      const int col  = cb * 16 + (lane & 15);
      const int rofs = (lane >> 4) << 3;
#pragma unroll
      for (int r = 0; r < 8; ++r) {
        size_t idx = (size_t)(rbase + r + rofs) * 128 + col;
        Kout[idx] = aK[r];
        Vout[idx] = aV[r];
      }
    }
  }
}

// ---------------------------------------------------------------------------
// Kernel 2: per-(batch, source) grouped attention + fc + residual/LN + merge.
// One 128-thread block per group; K/V group tiles staged in padded LDS.
// ---------------------------------------------------------------------------
__global__ __launch_bounds__(128) void attn_merge_kernel(
    const float* __restrict__ src, const unsigned char* __restrict__ mask,
    const float* __restrict__ Kg,  const float* __restrict__ Vg,
    const float* __restrict__ Wqs,
    const float* __restrict__ fcw, const float* __restrict__ fcb,
    const float* __restrict__ lng, const float* __restrict__ lnb,
    const float* __restrict__ m1w, const float* __restrict__ m1b,
    const float* __restrict__ m2w, const float* __restrict__ m2b,
    float* __restrict__ out) {
  __shared__ float Kl[64 * 129];
  __shared__ float Vl[64 * 129];
  __shared__ float srcv[128], qv[128], sc[128], ov[128], yv[128], h1v[128];
  __shared__ float red[128];

  const int g = blockIdx.x;            // g = b*32 + s;  kv rows = g*64 .. g*64+63
  const int b = g >> 5, s = g & 31;
  const int t = threadIdx.x;

  const size_t kvbase = (size_t)g * 64 * 128;
  for (int i = t; i < 64 * 128; i += 128) {
    int r = i >> 7, c = i & 127;
    Kl[r * 129 + c] = Kg[kvbase + i];
    Vl[r * 129 + c] = Vg[kvbase + i];
  }
  srcv[t] = src[(size_t)g * 128 + t];
  __syncthreads();

  // q = src_row @ Wqs^T
  {
    const float* wr = Wqs + (size_t)t * 128;
    float acc = 0.f;
    for (int k = 0; k < 128; ++k) acc += srcv[k] * wr[k];
    qv[t] = acc;
  }
  __syncthreads();

  // masked scaled scores: head h = t>>6, neighbor j = t&63
  {
    const int hh = t >> 6, j = t & 63;
    const float* kr = Kl + j * 129 + hh * 64;
    const float* qr = qv + hh * 64;
    float acc = 0.f;
#pragma unroll 8
    for (int i = 0; i < 64; ++i) acc += qr[i] * kr[i];
    acc *= 0.125f;                                   // / sqrt(DK=64)
    if (mask[(size_t)b * 2048 + s * 64 + j]) acc = -1e10f;
    sc[t] = acc;
  }
  __syncthreads();

  // softmax over 64 neighbors per head
  red[t] = sc[t]; __syncthreads();
  for (int off = 32; off >= 1; off >>= 1) {
    if ((t & 63) < off) red[t] = fmaxf(red[t], red[t + off]);
    __syncthreads();
  }
  const float mx = red[t & 64];
  __syncthreads();
  const float e = __expf(sc[t] - mx);
  red[t] = e; __syncthreads();
  for (int off = 32; off >= 1; off >>= 1) {
    if ((t & 63) < off) red[t] += red[t + off];
    __syncthreads();
  }
  const float sm = red[t & 64];
  __syncthreads();
  sc[t] = e / sm;
  __syncthreads();

  // out = attn @ V   (head h, feature d)
  {
    const int hh = t >> 6, d = t & 63;
    const float* ar = sc + hh * 64;
    float acc = 0.f;
    for (int j = 0; j < 64; ++j) acc += ar[j] * Vl[j * 129 + hh * 64 + d];
    ov[t] = acc;
  }
  __syncthreads();

  // fc + residual
  {
    const float* wr = fcw + (size_t)t * 128;
    float acc = fcb[t];
    for (int k = 0; k < 128; ++k) acc += ov[k] * wr[k];
    qv[t] = acc + srcv[t];                           // reuse qv as x
  }
  __syncthreads();

  // LayerNorm over 128
  red[t] = qv[t]; __syncthreads();
  for (int off = 64; off >= 1; off >>= 1) {
    if (t < off) red[t] += red[t + off];
    __syncthreads();
  }
  const float mu = red[0] * (1.f / 128.f);
  __syncthreads();
  const float dv = qv[t] - mu;
  red[t] = dv * dv; __syncthreads();
  for (int off = 64; off >= 1; off >>= 1) {
    if (t < off) red[t] += red[t + off];
    __syncthreads();
  }
  const float var = red[0] * (1.f / 128.f);
  yv[t] = dv * rsqrtf(var + 1e-5f) * lng[t] + lnb[t];
  __syncthreads();

  // MergeLayer: relu(cat([y, src]) @ m1w^T + b1) @ m2w^T + b2
  {
    const float* wr = m1w + (size_t)t * 256;
    float acc = m1b[t];
    for (int k = 0; k < 128; ++k) acc += yv[k] * wr[k];
    for (int k = 0; k < 128; ++k) acc += srcv[k] * wr[128 + k];
    h1v[t] = fmaxf(acc, 0.f);
  }
  __syncthreads();
  {
    const float* wr = m2w + (size_t)t * 128;
    float acc = m2b[t];
    for (int k = 0; k < 128; ++k) acc += h1v[k] * wr[k];
    out[(size_t)g * 128 + t] = acc;
  }
}

// ---------------------------------------------------------------------------
extern "C" void kernel_launch(void* const* d_in, const int* in_sizes, int n_in,
                              void* d_out, int out_size, void* d_ws,
                              size_t ws_size, hipStream_t stream) {
  const float* src = (const float*)d_in[0];
  const float* ngh = (const float*)d_in[1];
  const float* hid = (const float*)d_in[2];
  const unsigned char* mask = (const unsigned char*)d_in[3];  // numpy bool = u8
  const float* wih = (const float*)d_in[4];
  const float* whh = (const float*)d_in[5];
  const float* bih = (const float*)d_in[6];
  const float* bhh = (const float*)d_in[7];
  const float* wqs = (const float*)d_in[8];
  const float* wks = (const float*)d_in[9];
  const float* wvs = (const float*)d_in[10];
  const float* fcw = (const float*)d_in[11];
  const float* fcb = (const float*)d_in[12];
  const float* lng = (const float*)d_in[13];
  const float* lnb = (const float*)d_in[14];
  const float* m1w = (const float*)d_in[15];
  const float* m1b = (const float*)d_in[16];
  const float* m2w = (const float*)d_in[17];
  const float* m2b = (const float*)d_in[18];

  float* Kbuf = (float*)d_ws;                        // 256 MB
  float* Vbuf = Kbuf + (size_t)MROWS * 128;          // +256 MB (needs 512 MB ws)

  gru_kv_kernel<<<1024, 256, 0, stream>>>(ngh, hid, wih, whh, bih, bhh,
                                          wks, wvs, Kbuf, Vbuf);
  attn_merge_kernel<<<8192, 128, 0, stream>>>(src, mask, Kbuf, Vbuf, wqs,
                                              fcw, fcb, lng, lnb,
                                              m1w, m1b, m2w, m2b,
                                              (float*)d_out);
}
